// Group_16552803959356
// MI455X (gfx1250) — compile-verified, matrix operand order checked by reading, not run on hardware
//
#include <hip/hip_runtime.h>
#include <hip/hip_bf16.h>

// ---------------------------------------------------------------------------
// CDNA5 (gfx1250) implementation: all matmuls via v_wmma_f32_16x16x32_f16,
// plus gfx1250 async data movement:
//   - global_load_async_to_lds_b128 (ASYNCcnt) for attention q/k tile staging
//   - tensor_load_to_lds (TDM, TENSORcnt) with hardware LDS padding for the
//     token-major f16 A-tiles in the GEMM
// wave32 fragment layouts per CDNA5 ISA:
//   A (16x32 f16): lane L -> row M = L%16; elems 0..7 = K (L/16)*8 .. +7,
//                  elems 8..15 = K 16+(L/16)*8 .. +7   (two 16B chunks)
//   B (32x16 f16): same mapping with N = L%16 and LDS tile stored [N][K]
//   C/D (16x16 f32, 8 VGPRs): lane L -> col N = L%16; elem j -> row 8*(L/16)+j
// ---------------------------------------------------------------------------

typedef __attribute__((ext_vector_type(16))) _Float16 v16h;
typedef __attribute__((ext_vector_type(8)))  float    v8f;
typedef __attribute__((ext_vector_type(4)))  unsigned v4u;
typedef __attribute__((ext_vector_type(8)))  unsigned v8u;

#define T_TOK   32768   // 32*32*32 tokens
#define NHEAD   6
#define HDIM    32
#define CDIM    192
#define PDIM_   256

__device__ __forceinline__ v16h ld_frag(const _Float16* p) {
    // Two contiguous 8-half (16B) chunks at +0 and +16 halves.
    v16h f;
#pragma unroll
    for (int i = 0; i < 8; ++i) { f[i] = p[i]; f[i + 8] = p[i + 16]; }
    return f;
}

__device__ __forceinline__ v8f wmma_f16(v16h a, v16h b, v8f c) {
    return __builtin_amdgcn_wmma_f32_16x16x32_f16(
        false, a, false, b, (short)0, c, false, false);
}

__device__ __forceinline__ int win_token(int widx, int n) {
    // window widx in [0,64): (d0,h0,w0); position n in [0,512): (di,hi,wi)
    int d0 = widx >> 4, h0 = (widx >> 2) & 3, w0 = widx & 3;
    int di = n >> 6, hi = (n >> 3) & 7, wi = n & 7;
    return ((d0 * 8 + di) << 10) + ((h0 * 8 + hi) << 5) + (w0 * 8 + wi);
}

// low 32 bits of a generic pointer to __shared__ data == LDS byte address
__device__ __forceinline__ unsigned lds_addr32(const void* p) {
    return (unsigned)(unsigned long long)p;
}

// 16B global->LDS async copy (no VGPR round trip); tracked with ASYNCcnt.
__device__ __forceinline__ void async_copy_b128(const void* gsrc, unsigned lds) {
    asm volatile("global_load_async_to_lds_b128 %0, %1, off"
                 :: "v"(lds), "v"(gsrc) : "memory");
}
__device__ __forceinline__ void wait_asynccnt0() {
    asm volatile("s_wait_asynccnt 0x0" ::: "memory");
}

// TDM: 2D tile (rows x 64B) global->LDS with hardware padding of 8 halves
// per 32-half row (pad_interval code 3 = 16 DWORDs, pad_amount code 3 = 4
// DWORDs) -> reproduces the 40-half padded LDS stride. Issue once per wave.
__device__ __forceinline__ void tdm_load_tile_f16(
        const _Float16* gbase, unsigned lds, int row_stride_elems,
        int rows, int cols, int tensor_d0, int tensor_d1) {
    unsigned long long ga = (unsigned long long)gbase;
    v4u g0;
    g0[0] = 1u;                                  // count=1, user desc
    g0[1] = lds;                                 // lds_addr
    g0[2] = (unsigned)(ga & 0xFFFFFFFFull);      // global_addr[95:64]
    g0[3] = (unsigned)(ga >> 32) | (2u << 30);   // global_addr[120:96]|type=2
    v8u g1;
    g1[0] = (1u << 16)            // data_size = 2 bytes
          | (1u << 20)            // pad_enable
          | (3u << 22)            // pad_interval: 16 DWORDs (one 64B row)
          | (3u << 25);           // pad_amount: 4 DWORDs (8 halves)
    g1[1] = ((unsigned)tensor_d0 & 0xFFFFu) << 16;            // dim0[15:0]
    g1[2] = (((unsigned)tensor_d0 >> 16) & 0xFFFFu)
          | (((unsigned)tensor_d1 & 0xFFFFu) << 16);          // dim0 hi|dim1 lo
    g1[3] = (((unsigned)tensor_d1 >> 16) & 0xFFFFu)
          | (((unsigned)cols) << 16);                         // tile_dim0
    g1[4] = (unsigned)rows;                                   // tile_dim1
    g1[5] = (unsigned)row_stride_elems;                       // dim0_stride lo
    g1[6] = 0u;
    g1[7] = 0u;
    asm volatile("tensor_load_to_lds %0, %1" :: "s"(g0), "s"(g1) : "memory");
}
__device__ __forceinline__ void wait_tensorcnt0() {
    asm volatile("s_wait_tensorcnt 0x0" ::: "memory");
}

// ---------------------------------------------------------------------------
// Generic GEMM: Out[t][n] = epi( sum_k A[t][k] * W[n][k] + bias ), f16 out.
// A either channel-first f32 (A[k*T + t]) or token-major f16 (A[t*KD + k]).
// Tile: 128 tokens x 64 outputs, K-step 32. 8 waves: wave = M-fragment,
// each wave covers 4 N-fragments -> 1 A load + 4 B loads + 4 WMMA / K-step.
// Token-major A tiles are staged by the Tensor Data Mover.
// ---------------------------------------------------------------------------
template <bool A_CHANNEL_FIRST>
__global__ __launch_bounds__(256)
void gemm_kernel(const void* __restrict__ Aptr, const float* __restrict__ W,
                 const float* __restrict__ bias, _Float16* __restrict__ Out,
                 int KD, int ND, int do_gelu, int bias_mode) {
    __shared__ __align__(16) _Float16 lds_a[128 * 40];
    __shared__ __align__(16) _Float16 lds_b[64 * 40];

    const int tid  = threadIdx.x;
    const int wave = tid >> 5;
    const int lane = tid & 31;
    const int lrow = lane & 15;
    const int g    = lane >> 4;

    const int t0 = blockIdx.x * 128;
    const int n0 = blockIdx.y * 64;

    v8f acc[4];
#pragma unroll
    for (int nf = 0; nf < 4; ++nf)
#pragma unroll
        for (int j = 0; j < 8; ++j) acc[nf][j] = 0.0f;

    for (int k0 = 0; k0 < KD; k0 += 32) {
        __syncthreads();
        // stage A tile (128x32) as f16, layout [m][k], stride 40 halves
        if (A_CHANNEL_FIRST) {
            const float* A = (const float*)Aptr;
            for (int i = tid; i < 128 * 32; i += 256) {
                int m = i & 127, kk = i >> 7;
                lds_a[m * 40 + kk] =
                    (_Float16)A[(size_t)(k0 + kk) * T_TOK + (t0 + m)];
            }
        } else if (wave == 0) {
            // TDM DMA: 128 rows x 64B with row stride KD halves, HW-padded
            // into the 40-half LDS stride. Wave 0 issues; TENSORcnt drains
            // before the block barrier.
            const _Float16* A = (const _Float16*)Aptr;
            tdm_load_tile_f16(A + (size_t)t0 * KD + k0, lds_addr32(lds_a),
                              KD, 128, 32, KD - k0, T_TOK - t0);
        }
        // stage B tile (64x32) from weights [N][K], layout [n][k]
        for (int i = tid; i < 64 * 32; i += 256) {
            int n = i >> 5, kk = i & 31;
            lds_b[n * 40 + kk] = (_Float16)W[(size_t)(n0 + n) * KD + k0 + kk];
        }
        if (k0 + 32 < KD)  // prefetch next weight K-slab (global_prefetch_b8)
            __builtin_prefetch(&W[(size_t)(n0 + (tid >> 2)) * KD + k0 + 32], 0, 1);
        if (!A_CHANNEL_FIRST && wave == 0) wait_tensorcnt0();
        __syncthreads();

        v16h afrag = ld_frag(&lds_a[(wave * 16 + lrow) * 40 + g * 8]);
#pragma unroll
        for (int nf = 0; nf < 4; ++nf) {
            v16h bfrag = ld_frag(&lds_b[(nf * 16 + lrow) * 40 + g * 8]);
            acc[nf] = wmma_f16(afrag, bfrag, acc[nf]);
        }
    }

#pragma unroll
    for (int nf = 0; nf < 4; ++nf) {
        int ncol = n0 + nf * 16 + lrow;
        float bv;
        if (bias_mode == 1)  // kv bias: zeros for k half, v_b for v half
            bv = (ncol < CDIM) ? 0.0f : bias[ncol - CDIM];
        else
            bv = bias[ncol];
#pragma unroll
        for (int j = 0; j < 8; ++j) {
            int mrow = wave * 16 + g * 8 + j;
            float vv = acc[nf][j] + bv;
            if (do_gelu) vv = 0.5f * vv * (1.0f + erff(vv * 0.70710678118f));
            Out[(size_t)(t0 + mrow) * ND + ncol] = (_Float16)vv;
        }
    }
}

// ---------------------------------------------------------------------------
// Per-(token, head) L2 normalization of a 32-wide head vector, in place.
// One wave per row; lane = head-dim element; shuffle reduction.
// ---------------------------------------------------------------------------
__global__ __launch_bounds__(256)
void headnorm_kernel(_Float16* __restrict__ buf, int row_stride) {
    int wave = threadIdx.x >> 5, lane = threadIdx.x & 31;
    long row = (long)blockIdx.x * 8 + wave;   // over T_TOK * NHEAD rows
    int t = (int)(row / NHEAD), h = (int)(row % NHEAD);
    size_t idx = (size_t)t * row_stride + h * HDIM + lane;
    float v = (float)buf[idx];
    float s = v * v;
#pragma unroll
    for (int m = 1; m < 32; m <<= 1) s += __shfl_xor(s, m, 32);
    float inv = 1.0f / fmaxf(sqrtf(s), 1e-12f);
    buf[idx] = (_Float16)(v * inv);
}

// ---------------------------------------------------------------------------
// Windowed cosine attention, streaming over key tiles.
// Block = (window, head, 128-query tile). 8 waves; wave = query M-fragment.
// q/k tiles staged with global_load_async_to_lds_b128 (ASYNCcnt); V needs a
// transpose so it goes through VGPRs. Scores in [-scale, scale] -> fixed
// shift exp(s*scale - scale): single-pass softmax, no accumulator rescale.
// ---------------------------------------------------------------------------
__global__ __launch_bounds__(256)
void attn_kernel(const _Float16* __restrict__ q,   // [t][192] normalized
                 const _Float16* __restrict__ kv,  // [t][384]; k=0..191,v=192..
                 const float* __restrict__ logit_scale,
                 _Float16* __restrict__ out) {     // [t][192]
    __shared__ __align__(16) _Float16 lq[128 * 40];
    __shared__ __align__(16) _Float16 lk[64 * 40];
    __shared__ __align__(16) _Float16 lvt[32 * 72]; // V^T: [hd][key]
    __shared__ __align__(16) _Float16 lp[128 * 72]; // P:   [qrow][key]

    const int tid  = threadIdx.x;
    const int wave = tid >> 5;
    const int lane = tid & 31;
    const int lrow = lane & 15;
    const int g    = lane >> 4;

    const int widx = blockIdx.x;
    const int h    = blockIdx.y;
    const int q0   = blockIdx.z * 128;

    const float scale = __expf(fminf(logit_scale[h], 4.60517019f)); // ln(100)

    // stage normalized-q tile once: 512 16B chunks, async direct to LDS
#pragma unroll
    for (int r = 0; r < 2; ++r) {
        int i = r * 256 + tid;              // chunk id
        int m = i >> 2, c8 = (i & 3) * 8;
        int t = win_token(widx, q0 + m);
        async_copy_b128(q + (size_t)t * CDIM + h * HDIM + c8,
                        lds_addr32(&lq[m * 40 + c8]));
    }

    float den[8];
    v8f oacc[2];
#pragma unroll
    for (int j = 0; j < 8; ++j) den[j] = 0.0f;
#pragma unroll
    for (int nf = 0; nf < 2; ++nf)
#pragma unroll
        for (int j = 0; j < 8; ++j) oacc[nf][j] = 0.0f;

    for (int kt = 0; kt < 8; ++kt) {            // 8 x 64 = 512 keys
        __syncthreads();
        {   // k tile: 256 16B chunks, async direct to LDS
            int key = tid >> 2, c8 = (tid & 3) * 8;
            int t = win_token(widx, kt * 64 + key);
            async_copy_b128(kv + (size_t)t * (2 * CDIM) + h * HDIM + c8,
                            lds_addr32(&lk[key * 40 + c8]));
        }
        // v tile, transposed through VGPRs
        for (int i = tid; i < 64 * 32; i += 256) {
            int key = i >> 5, c = i & 31;
            int t = win_token(widx, kt * 64 + key);
            lvt[c * 72 + key] =
                kv[(size_t)t * (2 * CDIM) + CDIM + h * HDIM + c];
        }
        wait_asynccnt0();                   // covers lq (1st iter) + lk
        __syncthreads();

        // S = qn (16x32) x kn^T (32 x 64keys): one K=32 WMMA per N-fragment
        v16h aq = ld_frag(&lq[(wave * 16 + lrow) * 40 + g * 8]);
#pragma unroll
        for (int nf = 0; nf < 4; ++nf) {
            v16h bk = ld_frag(&lk[(nf * 16 + lrow) * 40 + g * 8]);
            v8f s;
#pragma unroll
            for (int j = 0; j < 8; ++j) s[j] = 0.0f;
            s = wmma_f16(aq, bk, s);
#pragma unroll
            for (int j = 0; j < 8; ++j) {
                float p = __expf(s[j] * scale - scale);
                den[j] += p;
                lp[(wave * 16 + g * 8 + j) * 72 + nf * 16 + lrow] = (_Float16)p;
            }
        }
        // P (16 x 64keys) x V (64keys x 32hd): 2 K-steps x 2 N-fragments
#pragma unroll
        for (int ks = 0; ks < 2; ++ks) {
            v16h ap = ld_frag(&lp[(wave * 16 + lrow) * 72 + ks * 32 + g * 8]);
#pragma unroll
            for (int nf = 0; nf < 2; ++nf) {
                v16h bv = ld_frag(&lvt[(nf * 16 + lrow) * 72 + ks * 32 + g * 8]);
                oacc[nf] = wmma_f16(ap, bv, oacc[nf]);
            }
        }
    }

    // denominator: reduce across the 16 lanes of each C-fragment group
    float dinv[8];
#pragma unroll
    for (int j = 0; j < 8; ++j) {
        float d = den[j];
        d += __shfl_xor(d, 1, 32);
        d += __shfl_xor(d, 2, 32);
        d += __shfl_xor(d, 4, 32);
        d += __shfl_xor(d, 8, 32);
        dinv[j] = 1.0f / d;
    }
#pragma unroll
    for (int nf = 0; nf < 2; ++nf)
#pragma unroll
        for (int j = 0; j < 8; ++j) {
            int m = wave * 16 + g * 8 + j;
            int t = win_token(widx, q0 + m);
            out[(size_t)t * CDIM + h * HDIM + nf * 16 + lrow] =
                (_Float16)(oacc[nf][j] * dinv[j]);
        }
}

// ---------------------------------------------------------------------------
// Fused output projection + LayerNorm + residual, writes channel-first f32.
// Tile: 32 tokens x full 192 outputs (rows complete -> LN in-block).
// 8 waves: wave w -> M-frag w&1, N-frags 3*(w>>1)..+2.
// ---------------------------------------------------------------------------
__global__ __launch_bounds__(256)
void proj_ln_kernel(const _Float16* __restrict__ ain,  // [t][192]
                    const float* __restrict__ W, const float* __restrict__ bias,
                    const float* __restrict__ x,       // channel-first f32
                    const float* __restrict__ ln_g, const float* __restrict__ ln_b,
                    float* __restrict__ out) {
    __shared__ __align__(16) _Float16 la[32 * 40];
    __shared__ __align__(16) _Float16 lb[192 * 40];
    __shared__ float prod[32 * 196];

    const int tid  = threadIdx.x;
    const int wave = tid >> 5;
    const int lane = tid & 31;
    const int lrow = lane & 15;
    const int g    = lane >> 4;

    const int t0 = blockIdx.x * 32;
    const int mf = wave & 1;
    const int nb = (wave >> 1) * 3;

    v8f acc[3];
#pragma unroll
    for (int c = 0; c < 3; ++c)
#pragma unroll
        for (int j = 0; j < 8; ++j) acc[c][j] = 0.0f;

    for (int k0 = 0; k0 < CDIM; k0 += 32) {
        __syncthreads();
        // A tile (32x32 f16 token-major): async 16B chunks -> LDS
        if (tid < 128) {
            int m = tid >> 2, c8 = (tid & 3) * 8;
            async_copy_b128(ain + (size_t)(t0 + m) * CDIM + k0 + c8,
                            lds_addr32(&la[m * 40 + c8]));
        }
        for (int i = tid; i < 192 * 32; i += 256) {
            int n = i >> 5, kk = i & 31;
            lb[n * 40 + kk] = (_Float16)W[(size_t)n * CDIM + k0 + kk];
        }
        wait_asynccnt0();
        __syncthreads();

        v16h afrag = ld_frag(&la[(mf * 16 + lrow) * 40 + g * 8]);
#pragma unroll
        for (int c = 0; c < 3; ++c) {
            v16h bfrag = ld_frag(&lb[((nb + c) * 16 + lrow) * 40 + g * 8]);
            acc[c] = wmma_f16(afrag, bfrag, acc[c]);
        }
    }

#pragma unroll
    for (int c = 0; c < 3; ++c) {
        int ncol = (nb + c) * 16 + lrow;
        float bv = bias[ncol];
#pragma unroll
        for (int j = 0; j < 8; ++j)
            prod[(mf * 16 + g * 8 + j) * 196 + ncol] = acc[c][j] + bv;
    }
    __syncthreads();

    // LayerNorm + residual: wave handles 4 tokens, lane covers 6 channels
    for (int it = 0; it < 4; ++it) {
        int tk = wave * 4 + it;
        float s = 0.0f, ss = 0.0f;
#pragma unroll
        for (int cc = 0; cc < 6; ++cc) {
            float v = prod[tk * 196 + lane + 32 * cc];
            s += v; ss += v * v;
        }
#pragma unroll
        for (int m = 1; m < 32; m <<= 1) {
            s  += __shfl_xor(s,  m, 32);
            ss += __shfl_xor(ss, m, 32);
        }
        float mean = s * (1.0f / CDIM);
        float var  = ss * (1.0f / CDIM) - mean * mean;
        float inv  = rsqrtf(var + 1e-5f);
        int t = t0 + tk;
#pragma unroll
        for (int cc = 0; cc < 6; ++cc) {
            int ch = lane + 32 * cc;
            float pv = prod[tk * 196 + ch];
            float ln = (pv - mean) * inv * ln_g[ch] + ln_b[ch];
            out[(size_t)ch * T_TOK + t] = x[(size_t)ch * T_TOK + t] + ln;
        }
    }
}

// ---------------------------------------------------------------------------
extern "C" void kernel_launch(void* const* d_in, const int* in_sizes, int n_in,
                              void* d_out, int out_size, void* d_ws, size_t ws_size,
                              hipStream_t stream) {
    const float* x       = (const float*)d_in[0];
    const float* prev_x  = (const float*)d_in[1];
    const float* fc1_w   = (const float*)d_in[2];
    const float* fc1_b   = (const float*)d_in[3];
    const float* fc2_w   = (const float*)d_in[4];
    const float* fc2_b   = (const float*)d_in[5];
    const float* q_w     = (const float*)d_in[6];
    const float* q_b     = (const float*)d_in[7];
    const float* kv_w    = (const float*)d_in[8];
    const float* v_b     = (const float*)d_in[9];
    const float* proj_w  = (const float*)d_in[10];
    const float* proj_b  = (const float*)d_in[11];
    const float* lscale  = (const float*)d_in[12];
    const float* ln_g    = (const float*)d_in[13];
    const float* ln_b    = (const float*)d_in[14];
    float* outp          = (float*)d_out;

    char* ws = (char*)d_ws;
    // hidden (16MB, reused for q after fc2) | pc 12MB | kv 24MB | attn_out 12MB
    _Float16* hidden = (_Float16*)(ws);                          // [T][256]
    _Float16* qbuf   = (_Float16*)(ws);                          // [T][192]
    _Float16* pc     = (_Float16*)(ws + (size_t)16 * 1024 * 1024); // [T][192]
    _Float16* kvbuf  = (_Float16*)(ws + (size_t)16 * 1024 * 1024
                                      + (size_t)T_TOK * CDIM * 2); // [T][384]
    _Float16* aout   = (_Float16*)((char*)kvbuf
                                      + (size_t)T_TOK * 2 * CDIM * 2); // [T][192]

    dim3 blk(256);
    // 1) hidden = GELU(prev_tok @ fc1_w^T + fc1_b)   (A: channel-first f32)
    gemm_kernel<true><<<dim3(T_TOK / 128, PDIM_ / 64), blk, 0, stream>>>(
        prev_x, fc1_w, fc1_b, hidden, PDIM_, PDIM_, /*gelu=*/1, /*bias=*/0);
    // 2) pc = hidden @ fc2_w^T + fc2_b               (A: token-major f16, TDM)
    gemm_kernel<false><<<dim3(T_TOK / 128, CDIM / 64), blk, 0, stream>>>(
        hidden, fc2_w, fc2_b, pc, PDIM_, CDIM, 0, 0);
    // 3) q = x_tok @ q_w^T + q_b  (overwrites hidden region; stream-ordered)
    gemm_kernel<true><<<dim3(T_TOK / 128, CDIM / 64), blk, 0, stream>>>(
        x, q_w, q_b, qbuf, CDIM, CDIM, 0, 0);
    // 4) kv = pc @ kv_w^T + [0; v_b]                 (A: token-major f16, TDM)
    gemm_kernel<false><<<dim3(T_TOK / 128, (2 * CDIM) / 64), blk, 0, stream>>>(
        pc, kv_w, v_b, kvbuf, CDIM, 2 * CDIM, 0, /*bias=*/1);
    // 5) cosine-normalize q and k per (token, head)
    headnorm_kernel<<<T_TOK * NHEAD / 8, blk, 0, stream>>>(qbuf, CDIM);
    headnorm_kernel<<<T_TOK * NHEAD / 8, blk, 0, stream>>>(kvbuf, 2 * CDIM);
    // 6) windowed attention (async q/k staging)
    attn_kernel<<<dim3(64, NHEAD, 512 / 128), blk, 0, stream>>>(
        qbuf, kvbuf, lscale, aout);
    // 7) proj + LN + residual -> channel-first f32 output
    proj_ln_kernel<<<T_TOK / 32, blk, 0, stream>>>(
        aout, proj_w, proj_b, x, ln_g, ln_b, outp);

    (void)in_sizes; (void)n_in; (void)out_size; (void)ws_size;
}
